// NonlocalBlock_68977174774048
// MI455X (gfx1250) — matrix operand order repaired and checked by reading, not compiled, not run on hardware
//
#include <hip/hip_runtime.h>

// Problem constants (from reference setup_inputs)
#define BATCH 4
#define CCH   64      // C
#define CI    32      // intermediate dim
#define NPIX  9216    // H*W = 96*96
#define MPOOL 4608    // NPIX / COMPRESSION

typedef __attribute__((ext_vector_type(16))) _Float16 v16h;
typedef __attribute__((ext_vector_type(8)))  _Float16 v8h;
typedef __attribute__((ext_vector_type(8)))  float    v8f;

// xor-butterfly via DS_SWIZZLE_B32 (group-of-32: [14:10]=xor, [9:5]=or, [4:0]=and)
#define SWZ_XOR(v, m) \
  __int_as_float(__builtin_amdgcn_ds_swizzle(__float_as_int(v), ((m) << 10) | 0x1f))

__device__ __forceinline__ unsigned pack2h(float a, float b) {
  union { _Float16 h[2]; unsigned u; } p;
  p.h[0] = (_Float16)a; p.h[1] = (_Float16)b;
  return p.u;
}

// ---------------------------------------------------------------------------
// WMMA operand loaders (16x16x32 f16, wave32; this file's contiguous K = 16*half + s order)
// ---------------------------------------------------------------------------
// A 16x32 f16, rows contiguous (ld = 32 halves):
//   lanes 0-15 : row = lane,    K = 0..15
//   lanes 16-31: row = lane-16, K = 16..31
__device__ __forceinline__ v16h load_mat_A(const _Float16* base, int lane) {
  int r = lane & 15;
  const _Float16* p = base + r * 32 + ((lane & 16) ? 16 : 0);
  v8h lo = *(const v8h*)(p);
  v8h hi = *(const v8h*)(p + 8);
  return __builtin_shufflevector(lo, hi, 0,1,2,3,4,5,6,7,8,9,10,11,12,13,14,15);
}

// B 32x16 f16, column n contiguous with leading dim ldcol:
//   lanes 0-15 : col = lane,    K = 0..15 ;  lanes 16-31: col = lane-16, K = 16..31
__device__ __forceinline__ v16h load_mat_B(const _Float16* col0, int ldcol, int lane) {
  int c = lane & 15;
  return *(const v16h*)(col0 + (size_t)c * ldcol + ((lane & 16) ? 16 : 0));
}

// ---------------------------------------------------------------------------
// Kernel 1: fused 1x1-conv projections + pair max-pool + f16 cast.
//   theta_h : [B][N][32]  row-major         (A operand of QK^T)
//   phi_h   : [B][M][32]  row-major         (B operand of QK^T)
//   g_t     : [B][32][M]  ci-major, M-index interleaved per 32-block
//             (j<16 -> 2j, else 2(j-16)+1) to match packed-P K ordering
// ---------------------------------------------------------------------------
__global__ __launch_bounds__(256) void proj_kernel(
    const float* __restrict__ x,
    const float* __restrict__ w_theta, const float* __restrict__ b_theta,
    const float* __restrict__ w_phi,   const float* __restrict__ b_phi,
    const float* __restrict__ w_g,     const float* __restrict__ b_g,
    _Float16* __restrict__ theta_h, _Float16* __restrict__ phi_h,
    _Float16* __restrict__ g_t)
{
  int idx = blockIdx.x * 256 + threadIdx.x;          // 0 .. B*N-1
  int b = idx / NPIX;
  int n = idx - b * NPIX;
  size_t xbase = (size_t)b * CCH * NPIX + n;

  float xr[CCH];
  #pragma unroll
  for (int c = 0; c < CCH; ++c) xr[c] = x[xbase + (size_t)c * NPIX];

  _Float16* tout = theta_h + (size_t)idx * CI;
  int m = n >> 1;
  _Float16* pout = phi_h + ((size_t)b * MPOOL + m) * CI;
  int j = m & 31;
  int pj = (j < 16) ? (j << 1) : (((j - 16) << 1) | 1);  // K-interleave for g_t
  size_t gpos = (size_t)(m & ~31) + pj;
  bool even = (n & 1) == 0;

  for (int ci = 0; ci < CI; ++ci) {
    float t = b_theta[ci], p = b_phi[ci], g = b_g[ci];
    const float* wt = w_theta + ci * CCH;   // uniform -> scalar loads
    const float* wp = w_phi   + ci * CCH;
    const float* wg = w_g     + ci * CCH;
    #pragma unroll
    for (int c = 0; c < CCH; ++c) {
      t = fmaf(xr[c], wt[c], t);
      p = fmaf(xr[c], wp[c], p);
      g = fmaf(xr[c], wg[c], g);
    }
    tout[ci] = (_Float16)t;
    float pm = fmaxf(p, SWZ_XOR(p, 1));    // pool pair (2m, 2m+1): adjacent lanes
    float gm = fmaxf(g, SWZ_XOR(g, 1));
    if (even) {
      pout[ci] = (_Float16)pm;
      g_t[((size_t)b * CI + ci) * MPOOL + gpos] = (_Float16)gm;
    }
  }
}

// w_out (64x32 f32) -> f16 with K(ci) interleaved to match packed-Y ordering.
__global__ void cvt_wout(const float* __restrict__ w, _Float16* __restrict__ wh) {
  int i = blockIdx.x * 256 + threadIdx.x;
  if (i < CCH * CI) {
    int c = i >> 5, ci = i & 31;
    int pci = (ci < 16) ? (ci << 1) : (((ci - 16) << 1) | 1);
    wh[c * 32 + pci] = (_Float16)w[i];
  }
}

// ---------------------------------------------------------------------------
// Kernel 2: fused flash-attention + output projection + residual.
// One wave owns 32 theta rows (two A tiles), streams M in tiles of 64.
// Online softmax: row-max via ds_swizzle butterfly; row-sum via P x ONES WMMA.
// P transposed D-layout -> A-layout through per-wave LDS (packed b32 stores).
// ---------------------------------------------------------------------------
__global__ __launch_bounds__(128) void attn_kernel(
    const _Float16* __restrict__ theta,
    const _Float16* __restrict__ phi,
    const _Float16* __restrict__ gt,
    const _Float16* __restrict__ wout,
    const float* __restrict__ b_out,
    const float* __restrict__ x,
    float* __restrict__ out)
{
  __shared__ _Float16 lds[4][4][16 * 32];   // [wave][P block][16x32 f16] = 16KB
  int lane = threadIdx.x & 31;
  int wid  = threadIdx.x >> 5;
  int gw   = blockIdx.x * 4 + wid;          // global wave id
  int b    = gw / (NPIX / 32);
  int tile = gw - b * (NPIX / 32);
  int R    = tile * 32;                     // first of 32 theta rows

  int colb = lane & 15;
  int rofs = (lane & 16) ? 8 : 0;           // D-layout row offset for this half

  v16h A[2];
  A[0] = load_mat_A(theta + ((size_t)b * NPIX + R) * CI, lane);
  A[1] = load_mat_A(theta + ((size_t)b * NPIX + R + 16) * CI, lane);

  v16h ONESB;                               // all-ones 32x16 B operand (row sums)
  #pragma unroll
  for (int i = 0; i < 16; ++i) ONESB[i] = (_Float16)1.0f;

  v8f  acc[2][2];                           // [A-tile][ci half], f32 accum of y
  float mrun[2][8], lrun[2][8];
  const v8f zC = {0.f,0.f,0.f,0.f,0.f,0.f,0.f,0.f};
  #pragma unroll
  for (int t = 0; t < 2; ++t) {
    acc[t][0] = zC; acc[t][1] = zC;
    #pragma unroll
    for (int d = 0; d < 8; ++d) { mrun[t][d] = -3.0e38f; lrun[t][d] = 0.f; }
  }

  const _Float16* phiB = phi + (size_t)b * MPOOL * CI;
  const _Float16* gB   = gt  + (size_t)b * CI * MPOOL;

  for (int cb = 0; cb < MPOOL; cb += 64) {
    __builtin_prefetch(phiB + (size_t)(cb + 64) * CI, 0, 1);

    v16h Bt0 = load_mat_B(phiB + (size_t)(cb +  0) * CI, CI, lane);
    v16h Bt1 = load_mat_B(phiB + (size_t)(cb + 16) * CI, CI, lane);
    v16h Bt2 = load_mat_B(phiB + (size_t)(cb + 32) * CI, CI, lane);
    v16h Bt3 = load_mat_B(phiB + (size_t)(cb + 48) * CI, CI, lane);

    #pragma unroll
    for (int t = 0; t < 2; ++t) {
      v8f S0 = __builtin_amdgcn_wmma_f32_16x16x32_f16(false, A[t], false, Bt0,
                                                      (short)0, zC, false, false);
      v8f S1 = __builtin_amdgcn_wmma_f32_16x16x32_f16(false, A[t], false, Bt1,
                                                      (short)0, zC, false, false);
      v8f S2 = __builtin_amdgcn_wmma_f32_16x16x32_f16(false, A[t], false, Bt2,
                                                      (short)0, zC, false, false);
      v8f S3 = __builtin_amdgcn_wmma_f32_16x16x32_f16(false, A[t], false, Bt3,
                                                      (short)0, zC, false, false);
      float sc[8];
      #pragma unroll
      for (int d = 0; d < 8; ++d) {
        // row max over 64 columns: elementwise over 4 tiles, then 16-lane butterfly
        float tm = fmaxf(fmaxf(S0[d], S1[d]), fmaxf(S2[d], S3[d]));
        tm = fmaxf(tm, SWZ_XOR(tm, 1));
        tm = fmaxf(tm, SWZ_XOR(tm, 2));
        tm = fmaxf(tm, SWZ_XOR(tm, 4));
        tm = fmaxf(tm, SWZ_XOR(tm, 8));
        float mn = fmaxf(mrun[t][d], tm);
        sc[d] = __expf(mrun[t][d] - mn);
        mrun[t][d] = mn;
        S0[d] = __expf(S0[d] - mn);
        S1[d] = __expf(S1[d] - mn);
        S2[d] = __expf(S2[d] - mn);
        S3[d] = __expf(S3[d] - mn);
      }
      // rescale y accumulators by exp(m_old - m_new) per row
      #pragma unroll
      for (int h = 0; h < 2; ++h)
        #pragma unroll
        for (int d = 0; d < 8; ++d)
          acc[t][h][d] *= sc[d];

      // stage P: packed b32 stores, K interleaved {c, c+16} -> halves {2c, 2c+1}
      unsigned* lp0 = (unsigned*)&lds[wid][t * 2 + 0][0];
      unsigned* lp1 = (unsigned*)&lds[wid][t * 2 + 1][0];
      #pragma unroll
      for (int d = 0; d < 8; ++d) {
        int row = d + rofs;
        lp0[row * 16 + colb] = pack2h(S0[d], S1[d]);
        lp1[row * 16 + colb] = pack2h(S2[d], S3[d]);
      }
      v16h P0 = load_mat_A(&lds[wid][t * 2 + 0][0], lane);
      v16h P1 = load_mat_A(&lds[wid][t * 2 + 1][0], lane);

      // row sums via WMMA against all-ones B (lands directly in stat layout)
      v8f Ls = __builtin_amdgcn_wmma_f32_16x16x32_f16(false, P0, false, ONESB,
                                                      (short)0, zC, false, false);
      Ls = __builtin_amdgcn_wmma_f32_16x16x32_f16(false, P1, false, ONESB,
                                                  (short)0, Ls, false, false);
      #pragma unroll
      for (int d = 0; d < 8; ++d)
        lrun[t][d] = lrun[t][d] * sc[d] + Ls[d];

      // PV: g_t rows stored in the same interleaved K order as packed P
      v16h G00 = load_mat_B(gB + cb,                   MPOOL, lane);
      v16h G01 = load_mat_B(gB + 16 * MPOOL + cb,      MPOOL, lane);
      v16h G10 = load_mat_B(gB + cb + 32,              MPOOL, lane);
      v16h G11 = load_mat_B(gB + 16 * MPOOL + cb + 32, MPOOL, lane);
      acc[t][0] = __builtin_amdgcn_wmma_f32_16x16x32_f16(false, P0, false, G00,
                                                         (short)0, acc[t][0], false, false);
      acc[t][0] = __builtin_amdgcn_wmma_f32_16x16x32_f16(false, P1, false, G10,
                                                         (short)0, acc[t][0], false, false);
      acc[t][1] = __builtin_amdgcn_wmma_f32_16x16x32_f16(false, P0, false, G01,
                                                         (short)0, acc[t][1], false, false);
      acc[t][1] = __builtin_amdgcn_wmma_f32_16x16x32_f16(false, P1, false, G11,
                                                         (short)0, acc[t][1], false, false);
    }
  }

  // normalize y = acc / l, stage as packed f16 [32 rows][32 ci-interleaved] in LDS
  unsigned* ly32 = (unsigned*)&lds[wid][0][0];
  #pragma unroll
  for (int t = 0; t < 2; ++t)
    #pragma unroll
    for (int d = 0; d < 8; ++d) {
      float inv = 1.0f / lrun[t][d];
      int row = t * 16 + d + rofs;
      ly32[row * 16 + colb] = pack2h(acc[t][0][d] * inv, acc[t][1][d] * inv);
    }

  // out = y @ w_out^T + b_out + x   (w_out_h stored with matching interleaved K)
  #pragma unroll
  for (int t = 0; t < 2; ++t) {
    v16h Y = load_mat_A(&lds[wid][0][0] + t * 16 * 32, lane);
    #pragma unroll
    for (int ct = 0; ct < 4; ++ct) {
      v16h W = load_mat_B(wout + ct * 16 * CI, CI, lane);
      v8f D = __builtin_amdgcn_wmma_f32_16x16x32_f16(false, Y, false, W,
                                                     (short)0, zC, false, false);
      int c = ct * 16 + colb;
      float bo = b_out[c];
      size_t base = ((size_t)b * CCH + c) * (size_t)NPIX + R + t * 16 + rofs;
      #pragma unroll
      for (int d = 0; d < 8; ++d) {
        size_t off = base + d;
        out[off] = D[d] + bo + x[off];
      }
    }
  }
}

// ---------------------------------------------------------------------------
extern "C" void kernel_launch(void* const* d_in, const int* in_sizes, int n_in,
                              void* d_out, int out_size, void* d_ws, size_t ws_size,
                              hipStream_t stream) {
  const float* x       = (const float*)d_in[0];
  const float* w_theta = (const float*)d_in[1];
  const float* b_theta = (const float*)d_in[2];
  const float* w_phi   = (const float*)d_in[3];
  const float* b_phi   = (const float*)d_in[4];
  const float* w_g     = (const float*)d_in[5];
  const float* b_g     = (const float*)d_in[6];
  const float* w_out   = (const float*)d_in[7];
  const float* b_out   = (const float*)d_in[8];
  float* out = (float*)d_out;

  // Workspace carve-up (f16 staging, ~4.7 MB, 32B-aligned chunks)
  _Float16* theta_h = (_Float16*)d_ws;                                // B*N*32
  _Float16* phi_h   = theta_h + (size_t)BATCH * NPIX * CI;            // B*M*32
  _Float16* g_t     = phi_h   + (size_t)BATCH * MPOOL * CI;           // B*32*M
  _Float16* w_out_h = g_t     + (size_t)BATCH * CI * MPOOL;           // 64*32

  proj_kernel<<<(BATCH * NPIX) / 256, 256, 0, stream>>>(
      x, w_theta, b_theta, w_phi, b_phi, w_g, b_g, theta_h, phi_h, g_t);
  cvt_wout<<<(CCH * CI + 255) / 256, 256, 0, stream>>>(w_out, w_out_h);
  attn_kernel<<<(BATCH * (NPIX / 32)) / 4, 128, 0, stream>>>(
      theta_h, phi_h, g_t, w_out_h, b_out, x, out);
}
